// DecoderFactoredLSTM_60189671686409
// MI455X (gfx1250) — compile-verified, matrix operand order checked by reading, not run on hardware
//
#include <hip/hip_runtime.h>
#include <stdint.h>
#include <stddef.h>

// ---------------- types ----------------
typedef __bf16 bf16_t;
typedef __attribute__((ext_vector_type(16))) __bf16 v16bf;
typedef __attribute__((ext_vector_type(8)))  float  v8f;
typedef int v4i_b __attribute__((vector_size(16)));   // matches builtin pointee

#define V_SZ 32000
#define E_SZ 512
#define H_SZ 1024
#define F_SZ 512
#define B_SZ 64
#define T_SZ 40
#define MB_SZ (T_SZ * B_SZ)   // 2560 rows for the batched chain GEMMs

// --- optional CDNA5 async global->LDS path (guarded; sync fallback) ---
#if defined(__has_builtin)
#if __has_builtin(__builtin_amdgcn_global_load_async_to_lds_b128)
#define HAS_ASYNC_LDS 1
#endif
#if __has_builtin(__builtin_amdgcn_s_wait_asynccnt)
#define HAS_WAIT_ASYNC 1
#endif
#endif
#ifndef HAS_ASYNC_LDS
#define HAS_ASYNC_LDS 0
#endif
#ifndef HAS_WAIT_ASYNC
#define HAS_WAIT_ASYNC 0
#endif

static __device__ __forceinline__ bf16_t f2bf(float f) {
  unsigned u = __builtin_bit_cast(unsigned, f);
  unsigned r = u + 0x7FFFu + ((u >> 16) & 1u);   // round-to-nearest-even
  unsigned short h = (unsigned short)(r >> 16);
  return __builtin_bit_cast(bf16_t, h);
}

static __device__ __forceinline__ float sigmoidf_(float x) {
  return 1.0f / (1.0f + __expf(-x));
}

static __device__ __forceinline__ v8f vzero8() {
  v8f z;
#pragma unroll
  for (int r = 0; r < 8; ++r) z[r] = 0.0f;
  return z;
}

// ---------------- fp32 -> bf16 weight conversion ----------------
__global__ void cvt_f32_to_bf16(const float* __restrict__ src,
                                bf16_t* __restrict__ dst, int n) {
  int i = blockIdx.x * blockDim.x + threadIdx.x;
  if (i < n) dst[i] = f2bf(src[i]);
}

// ---------------- embedding gather: captions[B,T] -> x[T,B,E] bf16 -------
__global__ void embed_gather(const float* __restrict__ emb,
                             const int* __restrict__ caps,
                             bf16_t* __restrict__ x, int total) {
  int i = blockIdx.x * blockDim.x + threadIdx.x;
  if (i >= total) return;
  int e  = i % E_SZ;
  int tb = i / E_SZ;
  int b  = tb % B_SZ;
  int t  = tb / B_SZ;
  int tok = caps[b * T_SZ + t];
  x[i] = f2bf(emb[(size_t)tok * E_SZ + e]);
}

// ---------------- zero LSTM state ----------------
__global__ void init_state(float* __restrict__ c, bf16_t* __restrict__ h, int n) {
  int i = blockIdx.x * blockDim.x + threadIdx.x;
  if (i < n) { c[i] = 0.0f; h[i] = f2bf(0.0f); }
}

// ---------------- generic TN GEMM: D[M,N] = A[M,K] * B[N,K]^T + bias -----
// block tile: 128 (M) x 128 (N); 8 waves, each 32x64 (2 m-subtiles x 4 n-subtiles)
// A tile (128x32) staged in LDS via async-to-LDS (ASYNCcnt) when available;
// B fragments streamed from global (K-contiguous rows) with prefetch.
__global__ __launch_bounds__(256) void gemm_tn_wmma(
    const bf16_t* __restrict__ A,   // [M,K]
    const bf16_t* __restrict__ Bm,  // [N,K]
    const float*  __restrict__ bias,// [N] or null
    float*        __restrict__ outF,// [M,N] or null
    bf16_t*       __restrict__ outB,// [M,N] or null
    int M, int N, int K) {
  __shared__ bf16_t sA[128 * 32];   // 8 KB

  const int tid  = threadIdx.x;
  const int lane = tid & 31;
  const int wave = tid >> 5;
  const int wm   = wave & 3;   // 0..3 -> 32-row subtile
  const int wn   = wave >> 2;  // 0..1 -> 64-col half
  const int blockM = blockIdx.y * 128;
  const int blockN = blockIdx.x * 128;
  const int lrow = lane & 15;
  const int lhi  = lane >> 4;  // 0: K 0..15 / M r ; 1: K 16..31 / M r+8

  v8f acc[2][4];
#pragma unroll
  for (int mt = 0; mt < 2; ++mt)
#pragma unroll
    for (int j = 0; j < 4; ++j) acc[mt][j] = vzero8();

  // A staging map: thread t copies 16 contiguous bf16 (32B): row = t/2, col = (t&1)*16
  const int aRow = tid >> 1;
  const int aCol = (tid & 1) * 16;

  for (int k0 = 0; k0 < K; k0 += 32) {
    const bf16_t* asrc = A + (size_t)(blockM + aRow) * K + k0 + aCol;
    bf16_t* adst = &sA[aRow * 32 + aCol];
#if HAS_ASYNC_LDS
    __builtin_amdgcn_global_load_async_to_lds_b128(
        (__attribute__((address_space(1))) v4i_b*)asrc,
        (__attribute__((address_space(3))) v4i_b*)adst, 0, 0);
    __builtin_amdgcn_global_load_async_to_lds_b128(
        (__attribute__((address_space(1))) v4i_b*)asrc,
        (__attribute__((address_space(3))) v4i_b*)adst, 16, 0);
#if HAS_WAIT_ASYNC
    __builtin_amdgcn_s_wait_asynccnt(0);
#else
    asm volatile("s_wait_asynccnt 0" ::: "memory");
#endif
#else
    ((uint4*)adst)[0] = ((const uint4*)asrc)[0];
    ((uint4*)adst)[1] = ((const uint4*)asrc)[1];
#endif
    __syncthreads();

    v16bf afrag[2];
#pragma unroll
    for (int mt = 0; mt < 2; ++mt)
      afrag[mt] = *(const v16bf*)(&sA[(wm * 32 + mt * 16 + lrow) * 32 + lhi * 16]);

#pragma unroll
    for (int j = 0; j < 4; ++j) {
      int n = blockN + wn * 64 + j * 16 + lrow;
      const bf16_t* bp = Bm + (size_t)n * K + k0 + lhi * 16;
      v16bf bfrag = *(const v16bf*)bp;
      __builtin_prefetch((const void*)(bp + 32), 0, 1);  // next K-slab
      acc[0][j] = __builtin_amdgcn_wmma_f32_16x16x32_bf16(
          false, afrag[0], false, bfrag, (short)0, acc[0][j], false, false);
      acc[1][j] = __builtin_amdgcn_wmma_f32_16x16x32_bf16(
          false, afrag[1], false, bfrag, (short)0, acc[1][j], false, false);
    }
    __syncthreads();
  }

#pragma unroll
  for (int mt = 0; mt < 2; ++mt) {
#pragma unroll
    for (int j = 0; j < 4; ++j) {
      int n = blockN + wn * 64 + j * 16 + lrow;
      float bv = bias ? bias[n] : 0.0f;
      int mBase = blockM + wm * 32 + mt * 16 + lhi * 8;
#pragma unroll
      for (int r = 0; r < 8; ++r) {
        float v = acc[mt][j][r] + bv;
        size_t o = (size_t)(mBase + r) * N + n;
        if (outF) outF[o] = v;
        if (outB) outB[o] = f2bf(v);
      }
    }
  }
}

// ---------------- fused LSTM step ----------------
// grid: 32 blocks * 8 waves = 256 wave-tiles = 4 (M=64/16) x 64 (H/16)
// each wave: 4 gate pre-activations for its 16x16 tile via WMMA over K=1024,
// then sigmoid/tanh cell update, writes c-state, next-h (bf16), hiddens slice.
__global__ __launch_bounds__(256) void lstm_step(
    const bf16_t* __restrict__ hprev,  // [B,H] bf16
    const bf16_t* __restrict__ Wi, const bf16_t* __restrict__ Wf,
    const bf16_t* __restrict__ Wo, const bf16_t* __restrict__ Wc, // [H,H]
    const float* __restrict__ bi, const float* __restrict__ bfv,
    const float* __restrict__ bo, const float* __restrict__ bc,   // [H]
    const float* __restrict__ ui, const float* __restrict__ uf,
    const float* __restrict__ uo, const float* __restrict__ uc,   // [B,H] slices
    float*  __restrict__ cstate,   // [B,H]
    bf16_t* __restrict__ hnext,    // [B,H]
    bf16_t* __restrict__ hout) {   // hiddens + t*B*H
  const int lane = threadIdx.x & 31;
  const int wave = threadIdx.x >> 5;
  const int tile = blockIdx.x * 8 + wave;  // 0..255
  const int m0 = (tile >> 6) * 16;         // 0..48
  const int n0 = (tile & 63) * 16;         // 0..1008
  const int lrow = lane & 15;
  const int lhi  = lane >> 4;

  v8f ai = vzero8(), af = vzero8(), ao = vzero8(), ac = vzero8();

  const bf16_t* ap = hprev + (size_t)(m0 + lrow) * H_SZ + lhi * 16;
  const size_t brow = (size_t)(n0 + lrow) * H_SZ + lhi * 16;

  for (int k0 = 0; k0 < H_SZ; k0 += 32) {
    v16bf afrag = *(const v16bf*)(ap + k0);
    v16bf bi_f = *(const v16bf*)(Wi + brow + k0);
    v16bf bf_f = *(const v16bf*)(Wf + brow + k0);
    v16bf bo_f = *(const v16bf*)(Wo + brow + k0);
    v16bf bc_f = *(const v16bf*)(Wc + brow + k0);
    ai = __builtin_amdgcn_wmma_f32_16x16x32_bf16(false, afrag, false, bi_f, (short)0, ai, false, false);
    af = __builtin_amdgcn_wmma_f32_16x16x32_bf16(false, afrag, false, bf_f, (short)0, af, false, false);
    ao = __builtin_amdgcn_wmma_f32_16x16x32_bf16(false, afrag, false, bo_f, (short)0, ao, false, false);
    ac = __builtin_amdgcn_wmma_f32_16x16x32_bf16(false, afrag, false, bc_f, (short)0, ac, false, false);
  }

  const int n = n0 + lrow;
  const float bvi = bi[n], bvf = bfv[n], bvo = bo[n], bvc = bc[n];
#pragma unroll
  for (int r = 0; r < 8; ++r) {
    int m = m0 + lhi * 8 + r;
    size_t idx = (size_t)m * H_SZ + n;
    float it = sigmoidf_(ui[idx] + bvi + ai[r]);
    float ft = sigmoidf_(uf[idx] + bvf + af[r]);
    float ot = sigmoidf_(uo[idx] + bvo + ao[r]);
    float ct = tanhf(uc[idx] + bvc + ac[r]);
    float cn = ft * cstate[idx] + it * ct;
    float h  = ot * cn;                 // reference: h = o_t * c (no tanh)
    cstate[idx] = cn;
    bf16_t hb = f2bf(h);
    hnext[idx] = hb;
    hout[idx]  = hb;
  }
}

// ---------------- host ----------------
extern "C" void kernel_launch(void* const* d_in, const int* in_sizes, int n_in,
                              void* d_out, int out_size, void* d_ws, size_t ws_size,
                              hipStream_t stream) {
  const float* B_emb = (const float*)d_in[0];
  const float *Vw[4], *Vb[4], *Sw[4], *Sb[4], *Uw[4], *Ub[4], *Ww[4], *Wb[4];
  for (int g = 0; g < 4; ++g) {
    int base = 1 + g * 8;
    Vw[g] = (const float*)d_in[base + 0]; Vb[g] = (const float*)d_in[base + 1];
    Sw[g] = (const float*)d_in[base + 2]; Sb[g] = (const float*)d_in[base + 3];
    Uw[g] = (const float*)d_in[base + 4]; Ub[g] = (const float*)d_in[base + 5];
    Ww[g] = (const float*)d_in[base + 6]; Wb[g] = (const float*)d_in[base + 7];
  }
  const float* Cw = (const float*)d_in[33];
  const float* Cb = (const float*)d_in[34];
  const int*  caps = (const int*)d_in[35];

  // workspace carve-out
  char* ws = (char*)d_ws;
  size_t off = 0;
  auto alloc = [&](size_t bytes) -> void* {
    void* p = ws + off;
    off += (bytes + 255) & ~(size_t)255;
    return p;
  };

  bf16_t* x_bf = (bf16_t*)alloc((size_t)MB_SZ * E_SZ * 2);
  bf16_t* v_bf = (bf16_t*)alloc((size_t)MB_SZ * F_SZ * 2);
  bf16_t* s_bf = (bf16_t*)alloc((size_t)MB_SZ * F_SZ * 2);
  float*  u[4];
  for (int g = 0; g < 4; ++g) u[g] = (float*)alloc((size_t)MB_SZ * H_SZ * 4);
  bf16_t *wV[4], *wS[4], *wU[4], *wW[4];
  for (int g = 0; g < 4; ++g) {
    wV[g] = (bf16_t*)alloc((size_t)F_SZ * E_SZ * 2);
    wS[g] = (bf16_t*)alloc((size_t)F_SZ * F_SZ * 2);
    wU[g] = (bf16_t*)alloc((size_t)H_SZ * F_SZ * 2);
    wW[g] = (bf16_t*)alloc((size_t)H_SZ * H_SZ * 2);
  }
  bf16_t* wC      = (bf16_t*)alloc((size_t)V_SZ * H_SZ * 2);
  bf16_t* hiddens = (bf16_t*)alloc((size_t)MB_SZ * H_SZ * 2);
  bf16_t* hs[2];
  hs[0] = (bf16_t*)alloc((size_t)B_SZ * H_SZ * 2);
  hs[1] = (bf16_t*)alloc((size_t)B_SZ * H_SZ * 2);
  float* cst = (float*)alloc((size_t)B_SZ * H_SZ * 4);

  auto cvt = [&](const float* s, bf16_t* d, size_t n) {
    cvt_f32_to_bf16<<<(unsigned)((n + 255) / 256), 256, 0, stream>>>(s, d, (int)n);
  };
  for (int g = 0; g < 4; ++g) {
    cvt(Vw[g], wV[g], (size_t)F_SZ * E_SZ);
    cvt(Sw[g], wS[g], (size_t)F_SZ * F_SZ);
    cvt(Uw[g], wU[g], (size_t)H_SZ * F_SZ);
    cvt(Ww[g], wW[g], (size_t)H_SZ * H_SZ);
  }
  cvt(Cw, wC, (size_t)V_SZ * H_SZ);

  // embedding gather -> x[T,B,E] bf16
  {
    int total = MB_SZ * E_SZ;
    embed_gather<<<(total + 255) / 256, 256, 0, stream>>>(B_emb, caps, x_bf, total);
  }

  // per-gate chain: v = x@Vw^T+b ; s = v@Sw^T+b ; u = s@Uw^T+b
  for (int g = 0; g < 4; ++g) {
    gemm_tn_wmma<<<dim3(F_SZ / 128, MB_SZ / 128), 256, 0, stream>>>(
        x_bf, wV[g], Vb[g], nullptr, v_bf, MB_SZ, F_SZ, E_SZ);
    gemm_tn_wmma<<<dim3(F_SZ / 128, MB_SZ / 128), 256, 0, stream>>>(
        v_bf, wS[g], Sb[g], nullptr, s_bf, MB_SZ, F_SZ, F_SZ);
    gemm_tn_wmma<<<dim3(H_SZ / 128, MB_SZ / 128), 256, 0, stream>>>(
        s_bf, wU[g], Ub[g], u[g], nullptr, MB_SZ, H_SZ, F_SZ);
  }

  // recurrence
  init_state<<<(B_SZ * H_SZ + 255) / 256, 256, 0, stream>>>(cst, hs[0], B_SZ * H_SZ);
  for (int t = 0; t < T_SZ; ++t) {
    size_t uoff = (size_t)t * B_SZ * H_SZ;
    lstm_step<<<32, 256, 0, stream>>>(
        hs[t & 1], wW[0], wW[1], wW[2], wW[3],
        Wb[0], Wb[1], Wb[2], Wb[3],
        u[0] + uoff, u[1] + uoff, u[2] + uoff, u[3] + uoff,
        cst, hs[(t + 1) & 1], hiddens + uoff);
  }

  // output projection: out[2560,32000] = hiddens @ C^T + Cb
  gemm_tn_wmma<<<dim3(V_SZ / 128, MB_SZ / 128), 256, 0, stream>>>(
      hiddens, wC, Cb, (float*)d_out, nullptr, MB_SZ, V_SZ, H_SZ);
}
